// ResidualAttention_55465207660581
// MI455X (gfx1250) — matrix operand order repaired and checked by reading, not compiled, not run on hardware
//
#include <hip/hip_runtime.h>
#include <math.h>
#include <stdint.h>

typedef __bf16 bf16;
typedef __attribute__((ext_vector_type(16))) __bf16 v16bf;
typedef __attribute__((ext_vector_type(8)))  __bf16 v8bf;
typedef __attribute__((ext_vector_type(8)))  float  v8f;

#define B_SZ  2
#define T_SZ  2048
#define D_SZ  2048
#define H_SZ  16
#define HD_SZ 128
#define BT_SZ (B_SZ * T_SZ)
#define QKV_ST (3 * D_SZ)                 /* fused QKV row stride */
#define ATTN_SCALE 0.08838834764831845f   /* HD^-0.5 */
#define RSQRT_D    0.022097086912079608f  /* D^-0.5  */

static __device__ __forceinline__ v16bf make_v16(v8bf lo, v8bf hi) {
  v16bf r;
#pragma unroll
  for (int i = 0; i < 8; ++i) { r[i] = lo[i]; r[i + 8] = hi[i]; }
  return r;
}

// Async 16B global->LDS copy (per lane). CDNA5 GLOBAL_LOAD_ASYNC_TO_LDS_B128,
// GV addressing, tracked by ASYNCcnt.
static __device__ __forceinline__ void async_ld_b128(void* lds_ptr, const void* gptr) {
  uint32_t l = (uint32_t)(uintptr_t)lds_ptr;   // generic LDS ptr: low 32 bits = LDS offset
  uint64_t g = (uint64_t)(uintptr_t)gptr;
  asm volatile("global_load_async_to_lds_b128 %0, %1, off"
               :: "v"(l), "v"(g) : "memory");
}

static __device__ __forceinline__ void wait_async0() {
#if __has_builtin(__builtin_amdgcn_s_wait_asynccnt)
  __builtin_amdgcn_s_wait_asynccnt(0);
#else
  asm volatile("s_wait_asynccnt 0x0" ::: "memory");
#endif
}

// ---------------- fp32 -> bf16 convert ----------------
__global__ __launch_bounds__(256) void cvt_bf16_kernel(const float* __restrict__ s,
                                                       bf16* __restrict__ d, int n) {
  int i = blockIdx.x * 256 + threadIdx.x;
  int stride = gridDim.x * 256;
  for (; i < n; i += stride) d[i] = (bf16)s[i];
}

// ---------------- RMSNorm (eps outside sqrt, as reference) ----------------
__global__ __launch_bounds__(256) void rmsnorm_kernel(const float* __restrict__ x,
                                                      const float* __restrict__ w,
                                                      bf16* __restrict__ out) {
  int row = blockIdx.x;
  const float* xr = x + (size_t)row * D_SZ;
  int lane = threadIdx.x & 31, wave = threadIdx.x >> 5;
  float ss = 0.f;
  for (int i = threadIdx.x; i < D_SZ; i += 256) { float v = xr[i]; ss += v * v; }
#pragma unroll
  for (int o = 16; o > 0; o >>= 1) ss += __shfl_xor(ss, o, 32);
  __shared__ float red[8];
  if (lane == 0) red[wave] = ss;
  __syncthreads();
  float tot = 0.f;
#pragma unroll
  for (int i = 0; i < 8; ++i) tot += red[i];
  float norm = sqrtf(tot) * RSQRT_D;
  float inv  = 1.0f / (norm + 1e-8f);
  bf16* orow = out + (size_t)row * D_SZ;
  for (int i = threadIdx.x; i < D_SZ; i += 256) orow[i] = (bf16)(w[i] * xr[i] * inv);
}

// ---------------- GEMM pipeline step (compile-time buffer index) ----------------
// One K-step of 64: 2 A-frags, 8 B-frags, 8 WMMAs; prefetches next tile async.
template <int CUR>
static __device__ __forceinline__ void gemm_step(
    bool pre, int k0n, const bf16* Arow, const bf16* Wrow,
    bf16 (&As)[2][64][64], bf16 (&Ws)[2][128][64],
    int ar, int ac, int wr, int wc,
    int am, int kb, int kbB, int nn, int wn, v8f (&acc)[4]) {
  wait_async0();            // this wave's tile copies landed in LDS
  __syncthreads();          // all waves' copies visible; prev readers of other buf done
  if (pre) {
    async_ld_b128(&As[1 - CUR][ar][ac],      Arow + k0n);
    async_ld_b128(&As[1 - CUR][ar][ac + 8],  Arow + k0n + 8);
    async_ld_b128(&Ws[1 - CUR][wr][wc],      Wrow + k0n);
    async_ld_b128(&Ws[1 - CUR][wr][wc + 8],  Wrow + k0n + 8);
    async_ld_b128(&Ws[1 - CUR][wr][wc + 16], Wrow + k0n + 16);
    async_ld_b128(&Ws[1 - CUR][wr][wc + 24], Wrow + k0n + 24);
  }
#pragma unroll
  for (int kc = 0; kc < 64; kc += 32) {
    v16bf af = make_v16(*(const v8bf*)&As[CUR][am][kc + kb],
                        *(const v8bf*)&As[CUR][am][kc + 16 + kb]);
    v16bf bfrag[4];
#pragma unroll
    for (int j = 0; j < 4; ++j) {
      const bf16* wsrow = &Ws[CUR][wn * 64 + j * 16 + nn][kc];
      bfrag[j] = make_v16(*(const v8bf*)&wsrow[kb], *(const v8bf*)&wsrow[16 + kb]);
    }
#pragma unroll
    for (int j = 0; j < 4; ++j)
      acc[j] = __builtin_amdgcn_wmma_f32_16x16x32_bf16(false, af, false, bfrag[j],
                                                       (short)0, acc[j], false, false);
  }
}

// ---------------- C[M,N] = A[M,K] * W[N,K]^T  (bf16 WMMA, f32 acc) ----------------
// Block tile 64x128, 8 waves (4 M-sub x 2 N-sub of 64 cols), K-step 64,
// double-buffered LDS fed by async global->LDS copies, manually 2x-unrolled.
__global__ __launch_bounds__(256) void gemm_kernel(const bf16* __restrict__ A,
                                                   const bf16* __restrict__ W,
                                                   void* __restrict__ Cout,
                                                   int M, int N, int Kd, int c_f32) {
  __shared__ __align__(16) bf16 As[2][64][64];    // 16 KB
  __shared__ __align__(16) bf16 Ws[2][128][64];   // 32 KB
  int tid = threadIdx.x, lane = tid & 31, wave = tid >> 5;
  int wm = wave >> 1, wn = wave & 1;
  int m0 = blockIdx.x * 64, n0 = blockIdx.y * 128;
  int ar = tid >> 2, ac = (tid & 3) * 16;         // A tile: two 16B chunks / thread
  int wr = tid >> 1, wc = (tid & 1) * 32;         // W tile: four 16B chunks / thread
  int am  = wm * 16 + (lane & 15);                // A-frag row
  int kb  = (lane >> 4) * 8;                      // A-frag K base (ISA 16-bit A layout)
  int kbB = (lane >> 4) * 16;                     // unused; B-frags use kb (same k order as A)
  int nn  = lane & 15;
  const bf16* Arow = A + (size_t)(m0 + ar) * Kd + ac;
  const bf16* Wrow = W + (size_t)(n0 + wr) * Kd + wc;

  v8f acc[4] = {};

  // prologue: tile 0 -> buffer 0
  async_ld_b128(&As[0][ar][ac],      Arow);
  async_ld_b128(&As[0][ar][ac + 8],  Arow + 8);
  async_ld_b128(&Ws[0][wr][wc],      Wrow);
  async_ld_b128(&Ws[0][wr][wc + 8],  Wrow + 8);
  async_ld_b128(&Ws[0][wr][wc + 16], Wrow + 16);
  async_ld_b128(&Ws[0][wr][wc + 24], Wrow + 24);

  int steps = Kd >> 6;                            // always even here (Kd = 2048)
  for (int s = 0; s < steps; s += 2) {
    gemm_step<0>(true, (s + 1) << 6, Arow, Wrow, As, Ws,
                 ar, ac, wr, wc, am, kb, kbB, nn, wn, acc);
    gemm_step<1>((s + 2) < steps, (s + 2) << 6, Arow, Wrow, As, Ws,
                 ar, ac, wr, wc, am, kb, kbB, nn, wn, acc);
  }

  int row0 = m0 + wm * 16 + (lane >> 4) * 8;      // C/D layout: row = r + 8*(lane>>4)
  int col0 = n0 + wn * 64 + nn;
  if (c_f32) {
    float* C = (float*)Cout;
#pragma unroll
    for (int r = 0; r < 8; ++r)
#pragma unroll
      for (int j = 0; j < 4; ++j)
        C[(size_t)(row0 + r) * N + col0 + j * 16] = acc[j][r];
  } else {
    bf16* C = (bf16*)Cout;
#pragma unroll
    for (int r = 0; r < 8; ++r)
#pragma unroll
      for (int j = 0; j < 4; ++j)
        C[(size_t)(row0 + r) * N + col0 + j * 16] = (bf16)acc[j][r];
  }
}

// ---------------- attention pipeline step (compile-time buffer index) ----------------
template <int CUR>
static __device__ __forceinline__ void attn_step(
    int it, int nsteps, const bf16* Kp, const bf16* Vp, size_t base,
    bf16 (&Ks)[2][32][HD_SZ], bf16 (&Vt)[2][HD_SZ][32], bf16 (&Ps)[8][16][32],
    const v16bf (&qf)[4], v8f (&o)[8], float (&rm)[8], float (&rl)[8],
    int lane, int wave, int nn, int kb, int kbB, int lr, int lc, int q0) {
  int kk0 = it << 5;
  wait_async0();              // this wave's Ks[CUR] async copies landed
  __syncthreads();            // Ks[CUR]/Vt[CUR] published (syncthreads waits dscnt)

  v8bf nv0 = {}, nv1 = {};
  bool more = (it + 1 < nsteps);
  if (more) {                 // prefetch key-block it+1 while computing on it
    int kn = kk0 + 32;
    const bf16* krow = Kp + base + (size_t)(kn + lr) * QKV_ST + lc;
    const bf16* vrow = Vp + base + (size_t)(kn + lr) * QKV_ST + lc;
    async_ld_b128(&Ks[1 - CUR][lr][lc],     krow);
    async_ld_b128(&Ks[1 - CUR][lr][lc + 8], krow + 8);
    nv0 = *(const v8bf*)vrow;
    nv1 = *(const v8bf*)(vrow + 8);
  }

  // S = Q K^T  (two 16-key subtiles x four 32-d chunks)
  v8f s0 = {}; v8f s1 = {};
#pragma unroll
  for (int c = 0; c < 4; ++c) {
    v16bf bk0 = make_v16(*(const v8bf*)&Ks[CUR][nn][c * 32 + kb],
                         *(const v8bf*)&Ks[CUR][nn][c * 32 + 16 + kb]);
    s0 = __builtin_amdgcn_wmma_f32_16x16x32_bf16(false, qf[c], false, bk0, (short)0, s0, false, false);
    v16bf bk1 = make_v16(*(const v8bf*)&Ks[CUR][16 + nn][c * 32 + kb],
                         *(const v8bf*)&Ks[CUR][16 + nn][c * 32 + 16 + kb]);
    s1 = __builtin_amdgcn_wmma_f32_16x16x32_bf16(false, qf[c], false, bk1, (short)0, s1, false, false);
  }

  // online softmax (f32); rows live in half-waves per C/D layout
#pragma unroll
  for (int r = 0; r < 8; ++r) {
    int qabs = q0 + (lane >> 4) * 8 + r;
    int ka = kk0 + nn, kb2 = kk0 + 16 + nn;
    float a0 = s0[r] * ATTN_SCALE + ((ka  <= qabs) ? 0.f : -1e9f);
    float a1 = s1[r] * ATTN_SCALE + ((kb2 <= qabs) ? 0.f : -1e9f);
    float mx = fmaxf(a0, a1);
#pragma unroll
    for (int t = 1; t < 16; t <<= 1) mx = fmaxf(mx, __shfl_xor(mx, t, 32));
    float nm   = fmaxf(rm[r], mx);
    float corr = __expf(rm[r] - nm);
    float p0 = __expf(a0 - nm);
    float p1 = __expf(a1 - nm);
    float rs = p0 + p1;
#pragma unroll
    for (int t = 1; t < 16; t <<= 1) rs += __shfl_xor(rs, t, 32);
    rl[r] = rl[r] * corr + rs;
    rm[r] = nm;
#pragma unroll
    for (int n = 0; n < 8; ++n) o[n][r] *= corr;
    int pr = (lane >> 4) * 8 + r;
    Ps[wave][pr][nn]      = (bf16)p0;   // per-wave region: same-wave LDS order
    Ps[wave][pr][16 + nn] = (bf16)p1;   // suffices, no barrier
  }

  // O += P V
  v16bf pf = make_v16(*(const v8bf*)&Ps[wave][nn][kb],
                      *(const v8bf*)&Ps[wave][nn][16 + kb]);
#pragma unroll
  for (int n = 0; n < 8; ++n) {
    v16bf vf = make_v16(*(const v8bf*)&Vt[CUR][n * 16 + nn][kb],
                        *(const v8bf*)&Vt[CUR][n * 16 + nn][16 + kb]);
    o[n] = __builtin_amdgcn_wmma_f32_16x16x32_bf16(false, pf, false, vf, (short)0, o[n], false, false);
  }

  // late transpose-store of next V block (loads overlapped the whole body)
  if (more) {
#pragma unroll
    for (int i = 0; i < 8; ++i) {
      Vt[1 - CUR][lc + i][lr]     = nv0[i];
      Vt[1 - CUR][lc + 8 + i][lr] = nv1[i];
    }
  }
}

// ---------------- causal flash attention ----------------
// 128 queries/block (8 waves x 16 rows), 32-key steps, double-buffered K/V,
// async K staging, one barrier per step, manually 2x-unrolled pipeline.
__global__ __launch_bounds__(256) void attn_kernel(const bf16* __restrict__ Q,
                                                   const bf16* __restrict__ K,
                                                   const bf16* __restrict__ V,
                                                   bf16* __restrict__ O) {
  int qblk = blockIdx.x, h = blockIdx.y, b = blockIdx.z;
  int tid = threadIdx.x, lane = tid & 31, wave = tid >> 5;
  int nn  = lane & 15;
  int kb  = (lane >> 4) * 8;
  int kbB = (lane >> 4) * 16;
  const size_t base  = ((size_t)b * T_SZ) * QKV_ST + (size_t)h * HD_SZ;
  const size_t obase = ((size_t)b * T_SZ) * D_SZ  + (size_t)h * HD_SZ;
  int q0 = qblk * 128 + wave * 16;

  __shared__ __align__(16) bf16 Ks[2][32][HD_SZ];   // 16 KB, row-major
  __shared__ __align__(16) bf16 Vt[2][HD_SZ][32];   // 16 KB, transposed
  __shared__ __align__(16) bf16 Ps[8][16][32];      //  8 KB, per-wave

  v16bf qf[4];
  {
    const bf16* qrow = Q + base + (size_t)(q0 + nn) * QKV_ST;
#pragma unroll
    for (int c = 0; c < 4; ++c)
      qf[c] = make_v16(*(const v8bf*)&qrow[c * 32 + kb],
                       *(const v8bf*)&qrow[c * 32 + 16 + kb]);
  }

  v8f o[8] = {};
  float rm[8], rl[8];
#pragma unroll
  for (int r = 0; r < 8; ++r) { rm[r] = -3.0e38f; rl[r] = 0.f; }

  int lr = tid >> 3;            // 0..31 key row
  int lc = (tid & 7) * 16;      // 0..112
  int nsteps = ((qblk + 1) * 128) >> 5;   // multiple of 4, always even

  // prologue: stage key-block 0 into buffer 0
  {
    const bf16* krow = K + base + (size_t)lr * QKV_ST + lc;
    const bf16* vrow = V + base + (size_t)lr * QKV_ST + lc;
    async_ld_b128(&Ks[0][lr][lc],     krow);
    async_ld_b128(&Ks[0][lr][lc + 8], krow + 8);
    v8bf v0 = *(const v8bf*)vrow;
    v8bf v1 = *(const v8bf*)(vrow + 8);
#pragma unroll
    for (int i = 0; i < 8; ++i) { Vt[0][lc + i][lr] = v0[i]; Vt[0][lc + 8 + i][lr] = v1[i]; }
  }

  for (int it = 0; it < nsteps; it += 2) {
    attn_step<0>(it,     nsteps, K, V, base, Ks, Vt, Ps, qf, o, rm, rl,
                 lane, wave, nn, kb, kbB, lr, lc, q0);
    attn_step<1>(it + 1, nsteps, K, V, base, Ks, Vt, Ps, qf, o, rm, rl,
                 lane, wave, nn, kb, kbB, lr, lc, q0);
  }

#pragma unroll
  for (int r = 0; r < 8; ++r) {
    float inv = 1.0f / rl[r];
    int qabs = q0 + (lane >> 4) * 8 + r;
    bf16* orow = O + obase + (size_t)qabs * D_SZ;
#pragma unroll
    for (int n = 0; n < 8; ++n) orow[n * 16 + nn] = (bf16)(o[n][r] * inv);
  }
}

extern "C" void kernel_launch(void* const* d_in, const int* in_sizes, int n_in,
                              void* d_out, int out_size, void* d_ws, size_t ws_size,
                              hipStream_t stream) {
  (void)in_sizes; (void)n_in; (void)out_size; (void)ws_size;
  const float* x    = (const float*)d_in[0];
  /* d_in[1] = attn_mask: causal + (-1e9) reconstructed in-kernel */
  const float* wnrm = (const float*)d_in[2];
  const float* wq   = (const float*)d_in[3];
  const float* wk   = (const float*)d_in[4];
  const float* wv   = (const float*)d_in[5];
  const float* wo   = (const float*)d_in[6];

  char* ws = (char*)d_ws;
  const size_t MB = (size_t)1 << 20;        // total workspace use: 112 MB
  bf16* xn   = (bf16*)(ws);                 // [BT,D]        16 MB
  bf16* wqkv = (bf16*)(ws + 16 * MB);       // [3D,D] fused  24 MB (wq|wk|wv rows)
  bf16* wob  = (bf16*)(ws + 40 * MB);       // [D,D]          8 MB
  bf16* QKV  = (bf16*)(ws + 48 * MB);       // [BT,3D] fused 48 MB
  bf16* Ab   = (bf16*)(ws + 96 * MB);       // [BT,D]        16 MB

  cvt_bf16_kernel<<<2048, 256, 0, stream>>>(wq, wqkv, D_SZ * D_SZ);
  cvt_bf16_kernel<<<2048, 256, 0, stream>>>(wk, wqkv + (size_t)D_SZ * D_SZ, D_SZ * D_SZ);
  cvt_bf16_kernel<<<2048, 256, 0, stream>>>(wv, wqkv + (size_t)2 * D_SZ * D_SZ, D_SZ * D_SZ);
  cvt_bf16_kernel<<<2048, 256, 0, stream>>>(wo, wob, D_SZ * D_SZ);
  rmsnorm_kernel<<<BT_SZ, 256, 0, stream>>>(x, wnrm, xn);

  // fused QKV projection: [4096,2048] x [6144,2048]^T
  gemm_kernel<<<dim3(BT_SZ / 64, QKV_ST / 128), 256, 0, stream>>>(
      xn, wqkv, QKV, BT_SZ, QKV_ST, D_SZ, 0);

  attn_kernel<<<dim3(T_SZ / 128, H_SZ, B_SZ), 256, 0, stream>>>(
      QKV, QKV + D_SZ, QKV + 2 * D_SZ, Ab);

  gemm_kernel<<<dim3(BT_SZ / 64, D_SZ / 128), 256, 0, stream>>>(
      Ab, wob, d_out, BT_SZ, D_SZ, D_SZ, 1);
}
